// AttentionXL_29772713295987
// MI455X (gfx1250) — compile-verified, matrix operand order checked by reading, not run on hardware
//
#include <hip/hip_runtime.h>
#include <math.h>

#define CUR   1024
#define FULL  2048
#define BSZ   4
#define DM    1024
#define NH    16
#define DH    64
#define PREV  (FULL - CUR)
#define ATT_SCALE 0.125f   // 1/sqrt(64)

typedef __attribute__((ext_vector_type(16))) __bf16 v16bf;
typedef __attribute__((ext_vector_type(8)))  __bf16 v8bf;
typedef __attribute__((ext_vector_type(8)))  float  v8f;

static __device__ __forceinline__ v16bf cat8(v8bf lo, v8bf hi) {
  return __builtin_shufflevector(lo, hi, 0, 1, 2, 3, 4, 5, 6, 7,
                                         8, 9, 10, 11, 12, 13, 14, 15);
}

static __device__ __forceinline__ v8f wmma_bf16(v16bf a, v16bf b, v8f c) {
  // D(16x16,f32) = A(16x32,bf16) * B(32x16,bf16) + C  (V_WMMA_F32_16X16X32_BF16)
  return __builtin_amdgcn_wmma_f32_16x16x32_bf16(false, a, false, b, (short)0, c,
                                                 false, false);
}

// Per-lane bf16 fragment: two contiguous 8-element runs.
// element e<8 : K = base + 8*half + e ; e>=8 : K = base + 16 + 8*half + (e-8)
static __device__ __forceinline__ v16bf load_frag(const __bf16* p) {
  return cat8(*(const v8bf*)p, *(const v8bf*)(p + 16));
}

// ---------------------------------------------------------------------------
// Elementwise fp32 -> bf16
// ---------------------------------------------------------------------------
__global__ __launch_bounds__(256) void cvt_f32_bf16(
    const float* __restrict__ src, __bf16* __restrict__ dst, int n)
{
  const int i = blockIdx.x * 256 + threadIdx.x;
  if (i < n) dst[i] = (__bf16)src[i];
}

// src[rows][cols] fp32 -> dst[cols][rows] bf16
__global__ __launch_bounds__(256) void transpose_cvt_bf16(
    const float* __restrict__ src, __bf16* __restrict__ dst, int rows, int cols)
{
  const int i = blockIdx.x * 256 + threadIdx.x;
  if (i >= rows * cols) return;
  const int r = i / cols, c = i - r * cols;
  dst[(size_t)c * rows + r] = (__bf16)src[i];
}

// ---------------------------------------------------------------------------
// bf16 GEMM + bias, fp32 accumulate: C[M,N] = A[M,K] * B[K,N] + bias[N]
// A row-major bf16 [M][K]; BT = B^T row-major bf16 [N][K].
// One wave: 16x64 tile, K-step 32 (4 x V_WMMA_F32_16X16X32_BF16).
// MODE 0: store fp32 [M][N].  MODE 1: store bf16 [M][N].
// MODE 2: kv split store: col<DM -> kb16[M][DM] bf16 (k);
//         col>=DM -> valT[(b*NH+h)*DH+d][FULL] bf16 (val transposed, row=j*BSZ+b).
// ---------------------------------------------------------------------------
template <int MODE>
__global__ __launch_bounds__(256) void gemm_bf16(
    const __bf16* __restrict__ A, const __bf16* __restrict__ BT,
    const float* __restrict__ bias, void* __restrict__ Cout,
    void* __restrict__ Cout2, int M, int N, int K)
{
  const int lane   = threadIdx.x & 31;
  const int wave   = threadIdx.x >> 5;
  const int tilesN = N >> 6;
  const int tiles  = (M >> 4) * tilesN;
  const int wid    = blockIdx.x * 8 + wave;
  if (wid >= tiles) return;
  const int m0   = (wid / tilesN) << 4;
  const int n0   = (wid % tilesN) << 6;
  const int half = lane >> 4, l16 = lane & 15;

  const __bf16* Ap = A + (size_t)(m0 + l16) * K + 8 * half;
  const __bf16* Bp = BT + (size_t)(n0 + l16) * K + 8 * half;
  const size_t cs = (size_t)16 * K;   // 16 columns ahead in BT

  v8f acc0 = {}, acc1 = {}, acc2 = {}, acc3 = {};
  for (int kk = 0; kk < K; kk += 32) {
    v16bf a  = load_frag(Ap + kk);
    v16bf b0 = load_frag(Bp + kk);
    v16bf b1 = load_frag(Bp + cs + kk);
    v16bf b2 = load_frag(Bp + 2 * cs + kk);
    v16bf b3 = load_frag(Bp + 3 * cs + kk);
    acc0 = wmma_bf16(a, b0, acc0);
    acc1 = wmma_bf16(a, b1, acc1);
    acc2 = wmma_bf16(a, b2, acc2);
    acc3 = wmma_bf16(a, b3, acc3);
  }

  const float bb0 = bias[n0 + l16];
  const float bb1 = bias[n0 + 16 + l16];
  const float bb2 = bias[n0 + 32 + l16];
  const float bb3 = bias[n0 + 48 + l16];
#pragma unroll
  for (int r = 0; r < 8; ++r) {
    const int row = m0 + r + 8 * half;
    const float e0 = acc0[r] + bb0, e1 = acc1[r] + bb1;
    const float e2 = acc2[r] + bb2, e3 = acc3[r] + bb3;
    if (MODE == 0) {
      float* Cp = (float*)Cout + (size_t)row * N + n0 + l16;
      Cp[0] = e0; Cp[16] = e1; Cp[32] = e2; Cp[48] = e3;
    } else if (MODE == 1) {
      __bf16* Cp = (__bf16*)Cout + (size_t)row * N + n0 + l16;
      Cp[0] = (__bf16)e0; Cp[16] = (__bf16)e1;
      Cp[32] = (__bf16)e2; Cp[48] = (__bf16)e3;
    } else {
      const int j = row >> 2, b = row & 3;   // row = j*BSZ + b
      const float ev[4] = {e0, e1, e2, e3};
#pragma unroll
      for (int s = 0; s < 4; ++s) {
        const int col = n0 + 16 * s + l16;
        if (col < DM) {
          ((__bf16*)Cout)[(size_t)row * DM + col] = (__bf16)ev[s];
        } else {
          const int c = col - DM;            // h*64 + d
          ((__bf16*)Cout2)[((size_t)(b * NH + (c >> 6)) * DH + (c & 63)) * FULL + j]
              = (__bf16)ev[s];
        }
      }
    }
  }
}

// ---------------------------------------------------------------------------
// Fused Transformer-XL flash attention, bf16 matmuls / fp32 softmax.
// One wave per (b, h, 16-row i-tile); streams 32-col j-tiles:
//   4 content WMMAs ((q+u)·k^T), 6 position WMMAs (48-wide (q+v)·r^T window at
//   s_base = j0+CUR-16-i0 >= 0; rel-shift diagonal t = jj+15-ii via LDS),
//   fp32 online softmax (16-lane butterflies), 4 P·V WMMAs (P relaid to
//   A-layout through LDS in bf16).  No block barriers: LDS is wave-private
//   and same-wave DS ops are in-order on CDNA5.
// ---------------------------------------------------------------------------
__global__ __launch_bounds__(256) void flash_attn_bf16(
    const float*  __restrict__ qb,    // [CUR*BSZ][DM] fp32
    const __bf16* __restrict__ kb,    // [FULL*BSZ][DM] bf16 (k)
    const __bf16* __restrict__ valT,  // [(b*NH+h)*DH + d][FULL] bf16
    const __bf16* __restrict__ rb16,  // [FULL][DM] bf16
    const float*  __restrict__ u,     // [DM]
    const float*  __restrict__ vvec,  // [DM]
    __bf16* __restrict__ vecb)        // [CUR*BSZ][DM] bf16
{
  __shared__ float  pwin[8][16][52];  // position window (48 used)
  __shared__ __bf16 ptb[8][16][32];   // P tile (bf16), C-layout -> A-layout
  const int lane = threadIdx.x & 31;
  const int wave = threadIdx.x >> 5;
  const int wid  = blockIdx.x * 8 + wave;
  const int ti   = wid & 63;
  const int bh   = wid >> 6;
  const int b = bh >> 4, h = bh & 15;
  const int i0 = ti << 4;
  const int half = lane >> 4, l16 = lane & 15;
  const int hb = h * DH;

  // A fragments for content (q+u) and position (q+v): K=64 -> 2 steps.
  // fp32 q + fp32 u/v, single rounding to bf16.
  const float* Aqf = qb + ((size_t)(i0 + l16) * BSZ + b) * DM + hb;
  v16bf aU[2], aV[2];
#pragma unroll
  for (int s = 0; s < 2; ++s) {
#pragma unroll
    for (int e = 0; e < 8; ++e) {
      const int k1 = 32 * s + 8 * half + e;
      const int k2 = k1 + 16;
      aU[s][e]     = (__bf16)(Aqf[k1] + u[hb + k1]);
      aU[s][8 + e] = (__bf16)(Aqf[k2] + u[hb + k2]);
      aV[s][e]     = (__bf16)(Aqf[k1] + vvec[hb + k1]);
      aV[s][8 + e] = (__bf16)(Aqf[k2] + vvec[hb + k2]);
    }
  }

  float mrow[8], lrow[8];
#pragma unroll
  for (int r = 0; r < 8; ++r) { mrow[r] = -3.0e38f; lrow[r] = 0.f; }
  v8f o0 = {}, o1 = {}, o2 = {}, o3 = {};

  const size_t kcol  = (size_t)BSZ * DM;   // j -> j+1 in kb
  const size_t dstrd = (size_t)16 * FULL;  // d -> d+16 in valT
  const __bf16* BvBase = valT + (size_t)((b * NH + h) * DH + l16) * FULL + 8 * half;
  const int jtmax = (i0 + 15 + PREV) >> 5; // inclusive, 32-wide tiles

  for (int jt = 0; jt <= jtmax; ++jt) {
    const int j0 = jt << 5;

    // ---- content: columns j0+l16 (tile0), j0+16+l16 (tile1) ----
    const __bf16* Bk0 = kb + ((size_t)(j0 + l16) * BSZ + b) * DM + hb + 8 * half;
    const __bf16* Bk1 = Bk0 + 16 * kcol;
    // ---- position window columns, clamped (clamps feed masked entries) ----
    const int sbase = j0 + CUR - 16 - i0;  // >= 0 always
    int sa = sbase + l16;       if (sa > FULL - 1) sa = FULL - 1;
    int sb = sbase + 16 + l16;  if (sb > FULL - 1) sb = FULL - 1;
    int sc = sbase + 32 + l16;  if (sc > FULL - 1) sc = FULL - 1;
    const __bf16* Br0 = rb16 + (size_t)sa * DM + hb + 8 * half;
    const __bf16* Br1 = rb16 + (size_t)sb * DM + hb + 8 * half;
    const __bf16* Br2 = rb16 + (size_t)sc * DM + hb + 8 * half;

    v8f accC0 = {}, accC1 = {}, accP0 = {}, accP1 = {}, accP2 = {};
#pragma unroll
    for (int s = 0; s < 2; ++s) {
      const int kk = 32 * s;
      accC0 = wmma_bf16(aU[s], load_frag(Bk0 + kk), accC0);
      accC1 = wmma_bf16(aU[s], load_frag(Bk1 + kk), accC1);
      accP0 = wmma_bf16(aV[s], load_frag(Br0 + kk), accP0);
      accP1 = wmma_bf16(aV[s], load_frag(Br1 + kk), accP1);
      accP2 = wmma_bf16(aV[s], load_frag(Br2 + kk), accP2);
    }

    // ---- rel-shift diagonal select ----
#pragma unroll
    for (int r = 0; r < 8; ++r) {
      const int row = r + 8 * half;
      pwin[wave][row][l16]      = accP0[r];
      pwin[wave][row][16 + l16] = accP1[r];
      pwin[wave][row][32 + l16] = accP2[r];
    }

    // ---- scores + mask + fp32 online softmax ----
#pragma unroll
    for (int r = 0; r < 8; ++r) {
      const int ii = r + 8 * half;
      const int gi = i0 + ii;
      float sc0 = (accC0[r] + pwin[wave][ii][l16 + 15 - ii]) * ATT_SCALE;
      float sc1 = (accC1[r] + pwin[wave][ii][16 + l16 + 15 - ii]) * ATT_SCALE;
      if (j0 + l16 > gi + PREV)      sc0 = -1e20f;
      if (j0 + 16 + l16 > gi + PREV) sc1 = -1e20f;

      float tm = fmaxf(sc0, sc1);
      tm = fmaxf(tm, __shfl_xor(tm, 1, 32));
      tm = fmaxf(tm, __shfl_xor(tm, 2, 32));
      tm = fmaxf(tm, __shfl_xor(tm, 4, 32));
      tm = fmaxf(tm, __shfl_xor(tm, 8, 32));
      const float mnew = fmaxf(mrow[r], tm);
      const float fac  = expf(mrow[r] - mnew);
      mrow[r] = mnew;
      const float p0 = expf(sc0 - mnew);
      const float p1 = expf(sc1 - mnew);
      float sum = p0 + p1;
      sum += __shfl_xor(sum, 1, 32);
      sum += __shfl_xor(sum, 2, 32);
      sum += __shfl_xor(sum, 4, 32);
      sum += __shfl_xor(sum, 8, 32);
      lrow[r] = lrow[r] * fac + sum;
      o0[r] *= fac; o1[r] *= fac; o2[r] *= fac; o3[r] *= fac;
      ptb[wave][ii][l16]      = (__bf16)p0;
      ptb[wave][ii][16 + l16] = (__bf16)p1;
    }

    // ---- O += P @ V : K=32, one WMMA step, N=64 (4 col tiles) ----
    v16bf a = cat8(*(const v8bf*)&ptb[wave][l16][8 * half],
                   *(const v8bf*)&ptb[wave][l16][16 + 8 * half]);
    const __bf16* Bv = BvBase + j0;
    o0 = wmma_bf16(a, load_frag(Bv), o0);
    o1 = wmma_bf16(a, load_frag(Bv + dstrd), o1);
    o2 = wmma_bf16(a, load_frag(Bv + 2 * dstrd), o2);
    o3 = wmma_bf16(a, load_frag(Bv + 3 * dstrd), o3);
  }

  // ---- normalize, write vec (bf16 for the output projection) ----
#pragma unroll
  for (int r = 0; r < 8; ++r) {
    const float inv = 1.0f / lrow[r];
    __bf16* Cp = vecb + ((size_t)(i0 + r + 8 * half) * BSZ + b) * DM + hb + l16;
    Cp[0]  = (__bf16)(o0[r] * inv);
    Cp[16] = (__bf16)(o1[r] * inv);
    Cp[32] = (__bf16)(o2[r] * inv);
    Cp[48] = (__bf16)(o3[r] * inv);
  }
}

// ---------------------------------------------------------------------------
extern "C" void kernel_launch(void* const* d_in, const int* in_sizes, int n_in,
                              void* d_out, int out_size, void* d_ws, size_t ws_size,
                              hipStream_t stream)
{
  (void)in_sizes; (void)n_in; (void)out_size; (void)ws_size;
  const float* inputs  = (const float*)d_in[0];
  const float* pos_emb = (const float*)d_in[1];
  const float* full_in = (const float*)d_in[2];
  const float* u   = (const float*)d_in[3];
  const float* vv  = (const float*)d_in[4];
  const float* Wkv = (const float*)d_in[5];
  const float* bkv = (const float*)d_in[6];
  const float* Wq  = (const float*)d_in[7];
  const float* bq  = (const float*)d_in[8];
  const float* Wr  = (const float*)d_in[9];
  const float* br  = (const float*)d_in[10];
  const float* Wo  = (const float*)d_in[11];
  const float* bo  = (const float*)d_in[12];
  // d_in[13] = mask: recomputed analytically (j > i+PREV), not read.
  float* out = (float*)d_out;

  char* ws = (char*)d_ws;
  __bf16* in_bf   = (__bf16*)ws;                 ws += (size_t)CUR * BSZ * DM * 2;   //  8MB
  __bf16* full_bf = (__bf16*)ws;                 ws += (size_t)FULL * BSZ * DM * 2;  // 16MB
  __bf16* pos_bf  = (__bf16*)ws;                 ws += (size_t)FULL * DM * 2;        //  4MB
  __bf16* WkvT    = (__bf16*)ws;                 ws += (size_t)2 * DM * DM * 2;      //  4MB
  __bf16* WqT     = (__bf16*)ws;                 ws += (size_t)DM * DM * 2;          //  2MB
  __bf16* WrT     = (__bf16*)ws;                 ws += (size_t)DM * DM * 2;          //  2MB
  __bf16* WoT     = (__bf16*)ws;                 ws += (size_t)DM * DM * 2;          //  2MB
  float*  qb      = (float*)ws;                  ws += (size_t)CUR * BSZ * DM * 4;   // 16MB
  __bf16* rb16    = (__bf16*)ws;                 ws += (size_t)FULL * DM * 2;        //  4MB
  __bf16* kb16    = (__bf16*)ws;                 ws += (size_t)FULL * BSZ * DM * 2;  // 16MB
  __bf16* valT    = (__bf16*)ws;                 ws += (size_t)BSZ * NH * DH * FULL * 2; // 16MB
  __bf16* vecb    = (__bf16*)ws;                 ws += (size_t)CUR * BSZ * DM * 2;   //  8MB

  // ---- operand conversion ----
  int n;
  n = CUR * BSZ * DM;
  cvt_f32_bf16<<<(n + 255) / 256, 256, 0, stream>>>(inputs, in_bf, n);
  n = FULL * BSZ * DM;
  cvt_f32_bf16<<<(n + 255) / 256, 256, 0, stream>>>(full_in, full_bf, n);
  n = FULL * DM;
  cvt_f32_bf16<<<(n + 255) / 256, 256, 0, stream>>>(pos_emb, pos_bf, n);
  n = DM * 2 * DM;
  transpose_cvt_bf16<<<(n + 255) / 256, 256, 0, stream>>>(Wkv, WkvT, DM, 2 * DM);
  n = DM * DM;
  transpose_cvt_bf16<<<(n + 255) / 256, 256, 0, stream>>>(Wq, WqT, DM, DM);
  transpose_cvt_bf16<<<(n + 255) / 256, 256, 0, stream>>>(Wr, WrT, DM, DM);
  transpose_cvt_bf16<<<(n + 255) / 256, 256, 0, stream>>>(Wo, WoT, DM, DM);

  // ---- projections (bf16 WMMA, fp32 accumulate) ----
  gemm_bf16<2><<<(FULL * BSZ / 16) * (2 * DM / 64) / 8, 256, 0, stream>>>(
      full_bf, WkvT, bkv, kb16, valT, FULL * BSZ, 2 * DM, DM);
  gemm_bf16<0><<<(CUR * BSZ / 16) * (DM / 64) / 8, 256, 0, stream>>>(
      in_bf, WqT, bq, qb, nullptr, CUR * BSZ, DM, DM);
  gemm_bf16<1><<<(FULL / 16) * (DM / 64) / 8, 256, 0, stream>>>(
      pos_bf, WrT, br, rb16, nullptr, FULL, DM, DM);

  // ---- fused attention, one launch for all (b, h, i-tiles) ----
  flash_attn_bf16<<<(BSZ * NH * (CUR / 16)) / 8, 256, 0, stream>>>(
      qb, kb16, valT, rb16, u, vv, vecb);

  // ---- output projection ----
  gemm_bf16<0><<<(CUR * BSZ / 16) * (DM / 64) / 8, 256, 0, stream>>>(
      vecb, WoT, bo, out, nullptr, CUR * BSZ, DM, DM);
}
